// MultiPosConLoss_66546223284605
// MI455X (gfx1250) — matrix operand order repaired and checked by reading, not compiled, not run on hardware
//
#include <hip/hip_runtime.h>
#include <hip/hip_bf16.h>

// ---------------------------------------------------------------------------
// MultiPosConLoss (SupCon) for MI455X / gfx1250.
//   B=6144, D=256, N=2B=12288, T=0.1
//   loss_i = logsumexp_j(s_ij/T) - (1/P_i) * sum_{j in pos(i)} s_ij/T
//   out    = mean_i loss_i
// s_ij in [-1,1] (unit vectors) => s/T in [-10,10]: no online max needed.
// Compute-bound (77 GFLOP vs 12.6 MB input) => f16 WMMA 16x16x32, f32 accum.
// Per-row stats are associative over columns => column-split blocks that
// atomically accumulate into per-row global buffers; P_i = 2*hist[label_i].
// ---------------------------------------------------------------------------

typedef __attribute__((ext_vector_type(16))) _Float16 v16h;
typedef __attribute__((ext_vector_type(8)))  float    v8f;

#define B_ROWS   6144
#define D_DIM    256
#define N_ROWS   12288            // 2*B_ROWS
#define INV_T    10.0f
#define LOG2E_T  14.4269504089f   // 10 * log2(e)
#define LDA      264              // padded LDS row stride in halves
#define ROWS_PER_BLOCK 128        // 4 waves x two 16-row strips each
#define SPLIT    8                // column-space split per row group
#define COLS_PER_BLOCK (N_ROWS / SPLIT)   // 1536
#define NTILE    32
#define NUMTILES (COLS_PER_BLOCK / NTILE) // 48
#define THREADS  128

// ---- init: zero per-row stats, histogram, output --------------------------
__global__ void init_kernel(float* gExp, float* gPos, int* hist, float* out) {
    int i = blockIdx.x * 256 + threadIdx.x;
    if (i < N_ROWS) { gExp[i] = 0.0f; gPos[i] = 0.0f; }
    if (i < 512)    hist[i] = 0;
    if (i == 0)     out[0] = 0.0f;
}

__global__ void hist_kernel(const int* __restrict__ labels, int* __restrict__ hist) {
    int i = blockIdx.x * 256 + threadIdx.x;
    if (i < B_ROWS) atomicAdd(&hist[labels[i]], 1);
}

// ---- one wave32 per row: L2-normalize and cast to f16 ---------------------
__global__ __launch_bounds__(256)
void l2norm_f16_kernel(const float* __restrict__ o1,
                       const float* __restrict__ o2,
                       _Float16* __restrict__ emb)
{
    int row  = blockIdx.x * 8 + (threadIdx.x >> 5);
    int lane = threadIdx.x & 31;
    const float* src = (row < B_ROWS) ? (o1 + (size_t)row * D_DIM)
                                      : (o2 + (size_t)(row - B_ROWS) * D_DIM);
    const float4* s4 = (const float4*)src;
    float4 x0 = s4[lane * 2 + 0];
    float4 x1 = s4[lane * 2 + 1];
    float ss = x0.x*x0.x + x0.y*x0.y + x0.z*x0.z + x0.w*x0.w
             + x1.x*x1.x + x1.y*x1.y + x1.z*x1.z + x1.w*x1.w;
#pragma unroll
    for (int m = 1; m < 32; m <<= 1) ss += __shfl_xor(ss, m, 32);
    float scale = 1.0f / fmaxf(sqrtf(ss), 1e-12f);

    union { _Float16 h[8]; uint4 q; } u;
    u.h[0] = (_Float16)(x0.x * scale); u.h[1] = (_Float16)(x0.y * scale);
    u.h[2] = (_Float16)(x0.z * scale); u.h[3] = (_Float16)(x0.w * scale);
    u.h[4] = (_Float16)(x1.x * scale); u.h[5] = (_Float16)(x1.y * scale);
    u.h[6] = (_Float16)(x1.z * scale); u.h[7] = (_Float16)(x1.w * scale);
    ((uint4*)(emb + (size_t)row * D_DIM))[lane] = u.q;
}

// ---- main: WMMA Gram tiles + streaming softmax stats ----------------------
__global__ __launch_bounds__(THREADS)
void supcon_main_kernel(const _Float16* __restrict__ emb,
                        const int* __restrict__ labels,
                        float* __restrict__ gExp,
                        float* __restrict__ gPos)
{
    // Reused region: first A staging (2 passes of 64 rows), then 2 B buffers.
    __shared__ __align__(16) _Float16 sStage[64 * LDA];   // 33.8 KB
    __shared__ int sLab[2][NTILE];

    const int tid  = threadIdx.x;
    const int lane = tid & 31;
    const int wave = tid >> 5;            // 0..3
    const int hi   = lane >> 4;           // K-half (A/B), row-half (C)
    const int ln   = lane & 15;
    const int rowBase = (blockIdx.x / SPLIT) * ROWS_PER_BLOCK;
    const int colBase = (blockIdx.x % SPLIT) * COLS_PER_BLOCK;

    // ---- hoist A fragments for both strips (j-invariant), via LDS staging --
    // A 16x32 f16 layout: lane halves [0..7]=K(k0+8*hi..), [8..15]=K(+16).
    v16h aFrag[2][8];
#pragma unroll
    for (int st = 0; st < 2; ++st) {
        __syncthreads();
        for (int c = tid; c < 64 * 32; c += THREADS) {
            int r = c >> 5, kc = c & 31;
            uint4 v = *(const uint4*)(emb + (size_t)(rowBase + st * 64 + r) * D_DIM + kc * 8);
            *(uint4*)(&sStage[r * LDA + kc * 8]) = v;
        }
        __syncthreads();
#pragma unroll
        for (int k8 = 0; k8 < 8; ++k8) {
            union { v16h v; uint4 q[2]; } a;
            const _Float16* ap = &sStage[(wave * 16 + ln) * LDA + k8 * 32 + hi * 8];
            a.q[0] = *(const uint4*)(ap);
            a.q[1] = *(const uint4*)(ap + 16);
            aFrag[st][k8] = a.v;
        }
    }
    __syncthreads();   // done reading A from LDS; region becomes B buffers

    _Float16* sB0 = sStage;
    _Float16* sB1 = sStage + NTILE * LDA;

    // per-lane row labels: strip st of this wave = global strip (wave + 4*st)
    int rlab[2][8];
#pragma unroll
    for (int st = 0; st < 2; ++st)
#pragma unroll
        for (int r = 0; r < 8; ++r) {
            int g = rowBase + (wave + 4 * st) * 16 + hi * 8 + r;
            rlab[st][r] = labels[(g < B_ROWS) ? g : g - B_ROWS];
        }

    float accE[2][8], accP[2][8];
#pragma unroll
    for (int st = 0; st < 2; ++st)
#pragma unroll
        for (int r = 0; r < 8; ++r) { accE[st][r] = 0.0f; accP[st][r] = 0.0f; }

    // Stage a 32-column tile into buffer pb via ASYNCcnt-tracked global->LDS DMA.
    auto stage_async = [&](int pb, int jc) {
        _Float16* dst = pb ? sB1 : sB0;
        for (int c = tid; c < NTILE * 32; c += THREADS) {
            int n = c >> 5, kc = c & 31;
            unsigned ldsOff = (unsigned)(uintptr_t)(&dst[n * LDA + kc * 8]);
            unsigned long long ga =
                (unsigned long long)(uintptr_t)(emb + (size_t)(jc + n) * D_DIM + kc * 8);
            asm volatile("global_load_async_to_lds_b128 %0, %1, off"
                         :: "v"(ldsOff), "v"(ga) : "memory");
        }
        if (tid < NTILE) {
            int g = jc + tid;
            sLab[pb][tid] = labels[(g < B_ROWS) ? g : g - B_ROWS];
        }
    };

    // prologue: fill buffer 0
    stage_async(0, colBase);
    asm volatile("s_wait_asynccnt 0x0" ::: "memory");
    __syncthreads();

    int p = 0;
    for (int t = 0; t < NUMTILES; ++t) {
        int jc = colBase + t * NTILE;
        if (t + 1 < NUMTILES) stage_async(p ^ 1, jc + NTILE);   // overlap DMA

        const _Float16* bufp = p ? sB1 : sB0;
#pragma unroll
        for (int ns = 0; ns < 2; ++ns) {
            v8f c0 = {};
            v8f c1 = {};
#pragma unroll
            for (int k8 = 0; k8 < 8; ++k8) {
                // B 32x16 f16 layout: lane = col (mod 16), [0..7]=K(k0+8*hi..),
                // [8..15]=K(+16).  One B frag feeds two WMMAs.
                union { v16h v; uint4 q[2]; } b;
                const _Float16* bp = &bufp[(ns * 16 + ln) * LDA + k8 * 32 + hi * 8];
                b.q[0] = *(const uint4*)(bp);
                b.q[1] = *(const uint4*)(bp + 16);
                c0 = __builtin_amdgcn_wmma_f32_16x16x32_f16(
                         false, aFrag[0][k8], false, b.v, (short)0, c0, false, false);
                c1 = __builtin_amdgcn_wmma_f32_16x16x32_f16(
                         false, aFrag[1][k8], false, b.v, (short)0, c1, false, false);
            }
            // C layout: VGPR r -> row (r + 8*hi), col = ln.
            int cl = sLab[p][ns * 16 + ln];
#pragma unroll
            for (int r = 0; r < 8; ++r) {
                float s0 = c0[r], s1 = c1[r];
                accE[0][r] += exp2f(s0 * LOG2E_T);
                accE[1][r] += exp2f(s1 * LOG2E_T);
                accP[0][r] += (cl == rlab[0][r]) ? s0 : 0.0f;
                accP[1][r] += (cl == rlab[1][r]) ? s1 : 0.0f;
            }
        }
        asm volatile("s_wait_asynccnt 0x0" ::: "memory");
        __syncthreads();
        p ^= 1;
    }

    // reduce over the 16 lanes sharing each row-half, then per-row atomics
#pragma unroll
    for (int m = 1; m < 16; m <<= 1)
#pragma unroll
        for (int st = 0; st < 2; ++st)
#pragma unroll
            for (int r = 0; r < 8; ++r) {
                accE[st][r] += __shfl_xor(accE[st][r], m, 32);
                accP[st][r] += __shfl_xor(accP[st][r], m, 32);
            }
    if (ln == 0) {   // lane 0: rows +0..7, lane 16: rows +8..15
#pragma unroll
        for (int st = 0; st < 2; ++st)
#pragma unroll
            for (int r = 0; r < 8; ++r) {
                int row = rowBase + (wave + 4 * st) * 16 + hi * 8 + r;
                atomicAdd(&gExp[row], accE[st][r]);
                atomicAdd(&gPos[row], accP[st][r]);
            }
    }
}

// ---- finalize: loss_i = log(sumExp_i) - (INV_T*sumPos_i)/P_i ; mean -------
__global__ __launch_bounds__(256)
void finalize_kernel(const float* __restrict__ gExp,
                     const float* __restrict__ gPos,
                     const int* __restrict__ hist,
                     const int* __restrict__ labels,
                     float* __restrict__ out)
{
    int i    = blockIdx.x * 256 + threadIdx.x;
    int lane = threadIdx.x & 31;
    int wave = threadIdx.x >> 5;
    float v = 0.0f;
    if (i < N_ROWS) {
        int lab   = labels[(i < B_ROWS) ? i : i - B_ROWS];
        float cnt = 2.0f * (float)hist[lab];
        v = __logf(gExp[i]) - INV_T * gPos[i] / cnt;
    }
#pragma unroll
    for (int m = 1; m < 32; m <<= 1) v += __shfl_xor(v, m, 32);
    __shared__ float sw[8];
    if (lane == 0) sw[wave] = v;
    __syncthreads();
    if (threadIdx.x == 0) {
        float s = 0.0f;
        for (int w = 0; w < 8; ++w) s += sw[w];
        atomicAdd(out, s * (1.0f / (float)N_ROWS));
    }
}

extern "C" void kernel_launch(void* const* d_in, const int* in_sizes, int n_in,
                              void* d_out, int out_size, void* d_ws, size_t ws_size,
                              hipStream_t stream) {
    const float* o1     = (const float*)d_in[0];
    const float* o2     = (const float*)d_in[1];
    const int*   labels = (const int*)d_in[2];
    float*       out    = (float*)d_out;

    char* ws = (char*)d_ws;
    _Float16* emb  = (_Float16*)ws;                                   // 6,291,456 B
    float*    gExp = (float*)(ws + (size_t)N_ROWS * D_DIM * 2);       // 49,152 B
    float*    gPos = gExp + N_ROWS;                                   // 49,152 B
    int*      hist = (int*)(gPos + N_ROWS);                           // 2,048 B

    init_kernel<<<(N_ROWS + 255) / 256, 256, 0, stream>>>(gExp, gPos, hist, out);
    hist_kernel<<<(B_ROWS + 255) / 256, 256, 0, stream>>>(labels, hist);
    l2norm_f16_kernel<<<N_ROWS / 8, 256, 0, stream>>>(o1, o2, emb);
    supcon_main_kernel<<<(N_ROWS / ROWS_PER_BLOCK) * SPLIT, THREADS, 0, stream>>>(
        emb, labels, gExp, gPos);
    finalize_kernel<<<(N_ROWS + 255) / 256, 256, 0, stream>>>(gExp, gPos, hist, labels, out);
}
